// Head_87608742904077
// MI455X (gfx1250) — matrix operand order repaired and checked by reading, not compiled, not run on hardware
//
#include <hip/hip_runtime.h>

#define B_ 16
#define T_ 2048
#define C_ 384
#define D_ 64

typedef __attribute__((ext_vector_type(8)))  _Float16 v8h;
typedef __attribute__((ext_vector_type(16))) _Float16 v16h;
typedef __attribute__((ext_vector_type(8)))  float    v8f;
typedef __attribute__((ext_vector_type(4)))  float    v4f;

static __device__ __forceinline__ v16h cat16(v8h lo, v8h hi) {
  v16h r;
#pragma unroll
  for (int i = 0; i < 8; ++i) { r[i] = lo[i]; r[i + 8] = hi[i]; }
  return r;
}

// load 8 consecutive f32 and convert to 8 f16
static __device__ __forceinline__ v8h cvt8(const float* p) {
  v4f a = *(const v4f*)p;
  v4f b = *(const v4f*)(p + 4);
  v8h r;
#pragma unroll
  for (int i = 0; i < 4; ++i) { r[i] = (_Float16)a[i]; r[i + 4] = (_Float16)b[i]; }
  return r;
}

static __device__ __forceinline__ v8f zero8() {
  v8f z;
#pragma unroll
  for (int i = 0; i < 8; ++i) z[i] = 0.0f;
  return z;
}

// ---------------- kernel 1: W -> f16 transposed [3][64][384] ----------------
__global__ void prep_w(const float* __restrict__ Wq, const float* __restrict__ Wk,
                       const float* __restrict__ Wv, _Float16* __restrict__ wt) {
  int idx = blockIdx.x * blockDim.x + threadIdx.x;   // 3 * C_ * D_ total
  if (idx >= 3 * C_ * D_) return;
  int w   = idx / (C_ * D_);
  int rem = idx % (C_ * D_);
  int d = rem / C_;
  int c = rem % C_;
  const float* W = (w == 0) ? Wq : ((w == 1) ? Wk : Wv);
  wt[(size_t)w * D_ * C_ + (size_t)d * C_ + c] = (_Float16)W[c * D_ + d];
}

// ---------------- kernel 2: Q/K/V projections, one wave per 16 rows ----------
__global__ void __launch_bounds__(32)
proj_qkv(const float* __restrict__ x, const _Float16* __restrict__ wt,
         _Float16* __restrict__ qh, _Float16* __restrict__ kh,
         _Float16* __restrict__ vt) {
  const int lane = threadIdx.x;
  const int tilesPerB = T_ / 16;
  const int b  = blockIdx.x / tilesPerB;
  const int t0 = (blockIdx.x % tilesPerB) * 16;
  const int m  = lane & 15;
  const int h  = lane >> 4;

  v8f acc[3][4];
#pragma unroll
  for (int w = 0; w < 3; ++w)
#pragma unroll
    for (int nt = 0; nt < 4; ++nt) acc[w][nt] = zero8();

  const float* xrow = x + ((size_t)(b * T_ + t0 + m)) * C_;

  for (int c0 = 0; c0 < C_; c0 += 32) {
    if (c0 + 32 < C_) __builtin_prefetch(xrow + c0 + 32, 0, 0);
    // A fragment (16x32 f16): lanes 0-15 K{0..7,16..23}, lanes 16-31 K{8..15,24..31}
    v16h a = cat16(cvt8(xrow + c0 + h * 8), cvt8(xrow + c0 + 16 + h * 8));
#pragma unroll
    for (int w = 0; w < 3; ++w) {
      // batch all 4 B-fragment loads (distinct regs) before the 4 WMMAs
      v16h bb[4];
#pragma unroll
      for (int nt = 0; nt < 4; ++nt) {
        const _Float16* wp = wt + (size_t)w * D_ * C_ + (size_t)(nt * 16 + m) * C_ + c0 + h * 8;
        bb[nt] = cat16(*(const v8h*)wp, *(const v8h*)(wp + 16));
      }
#pragma unroll
      for (int nt = 0; nt < 4; ++nt)
        acc[w][nt] = __builtin_amdgcn_wmma_f32_16x16x32_f16(
            false, a, false, bb[nt], (short)0, acc[w][nt], false, false);
    }
  }

  // C layout: VGPR r -> row (t0 + r + 8h), col = nt*16 + m
#pragma unroll
  for (int nt = 0; nt < 4; ++nt) {
#pragma unroll
    for (int r = 0; r < 8; ++r) {
      const int row = t0 + r + 8 * h;
      const int col = nt * 16 + m;
      qh[((size_t)(b * T_ + row)) * D_ + col] = (_Float16)acc[0][nt][r];
      kh[((size_t)(b * T_ + row)) * D_ + col] = (_Float16)acc[1][nt][r];
      vt[((size_t)b * D_ + col) * T_ + row]   = (_Float16)acc[2][nt][r];  // V transposed
    }
  }
}

// ---------------- kernel 3: flash attention (transposed tiles) ---------------
// Computes S^T = K·Q^T and O^T = V^T·P^T so each softmax row (fixed t) lives in
// a single lane pair: per-lane scalar running stats, one shfl_xor(16) per
// reduction instead of 8 shuffles per row-group.
__global__ void __launch_bounds__(32)
attn(const _Float16* __restrict__ qh, const _Float16* __restrict__ kh,
     const _Float16* __restrict__ vt, float* __restrict__ out) {
  __shared__ __align__(16) _Float16 lds_p[16][32];   // [t-local][s-local]

  const int lane = threadIdx.x;
  const int tilesPerB = T_ / 16;
  const int b  = blockIdx.x / tilesPerB;
  const int t0 = (blockIdx.x % tilesPerB) * 16;
  const int m  = lane & 15;       // t-local column owned by this lane
  const int h  = lane >> 4;
  const int trow = t0 + m;        // global t row this lane reduces
  const float NEG_INF = -__builtin_inff();

  // Q^T B-fragments: lane column t = t0+m, K=d: lanes 0-15 d{0..7,16..23}, 16-31 d{8..15,24..31}
  const _Float16* qrow = qh + ((size_t)(b * T_ + trow)) * D_;
  v16h qb[2];
#pragma unroll
  for (int kk = 0; kk < 2; ++kk)
    qb[kk] = cat16(*(const v8h*)(qrow + kk * 32 + h * 8),
                   *(const v8h*)(qrow + kk * 32 + 16 + h * 8));

  v8f o[4];                        // O^T accumulators: (d-tile nt) x (16 t cols)
#pragma unroll
  for (int nt = 0; nt < 4; ++nt) o[nt] = zero8();
  float mrun = NEG_INF, lrun = 0.0f;

  const int nT = (t0 + 47) >> 5;   // causal: last KV tile covers t0+15
  for (int st = 0; st < nT; ++st) {
    const int s0 = st * 32;

    if (st + 1 < nT) {             // -> global_prefetch_b8
      __builtin_prefetch(kh + ((size_t)(b * T_ + s0 + 32 + m)) * D_, 0, 0);
      __builtin_prefetch(vt + ((size_t)b * D_ + m) * T_ + s0 + 32, 0, 0);
    }

    // ---- S^T = K_tile(32s x 64d) * Q^T : two 16-s subtiles ----
    v16h ka[2][2];                 // [s-subtile][k-step] A fragments of K
#pragma unroll
    for (int ss = 0; ss < 2; ++ss) {
      const _Float16* kr = kh + ((size_t)(b * T_ + s0 + ss * 16 + m)) * D_;
#pragma unroll
      for (int kk = 0; kk < 2; ++kk)
        ka[ss][kk] = cat16(*(const v8h*)(kr + kk * 32 + h * 8),
                           *(const v8h*)(kr + kk * 32 + 16 + h * 8));
    }
    v8f sc[2] = {zero8(), zero8()};
#pragma unroll
    for (int ss = 0; ss < 2; ++ss)
#pragma unroll
      for (int kk = 0; kk < 2; ++kk)
        sc[ss] = __builtin_amdgcn_wmma_f32_16x16x32_f16(
            false, ka[ss][kk], false, qb[kk], (short)0, sc[ss], false, false);

    // ---- mask + online softmax: column t per lane, s across registers ----
    float vv[2][8];
    float mx = NEG_INF;
#pragma unroll
    for (int ss = 0; ss < 2; ++ss)
#pragma unroll
      for (int r = 0; r < 8; ++r) {
        const int s = s0 + ss * 16 + r + 8 * h;
        float v = sc[ss][r] * 0.125f;
        // reference quirk: tril, then exact zeros -> -inf
        if (s > trow || v == 0.0f) v = NEG_INF;
        vv[ss][r] = v;
        mx = fmaxf(mx, v);
      }
    mx = fmaxf(mx, __shfl_xor(mx, 16, 32));     // combine the two s-halves
    const float mnew = fmaxf(mrun, mx);
    const float fac = (mrun == NEG_INF) ? 0.0f : __expf(mrun - mnew);
    float rs = 0.0f;
    float pv[2][8];
#pragma unroll
    for (int ss = 0; ss < 2; ++ss)
#pragma unroll
      for (int r = 0; r < 8; ++r) {
        const float e = (vv[ss][r] == NEG_INF) ? 0.0f : __expf(vv[ss][r] - mnew);
        pv[ss][r] = e;
        rs += e;
      }
    rs += __shfl_xor(rs, 16, 32);
    lrun = lrun * fac + rs;
    mrun = mnew;

#pragma unroll
    for (int nt = 0; nt < 4; ++nt)
#pragma unroll
      for (int r = 0; r < 8; ++r) o[nt][r] *= fac;

    // ---- P^T C-layout -> B-fragment layout via LDS ([t][s], s contiguous) ----
#pragma unroll
    for (int ss = 0; ss < 2; ++ss)
#pragma unroll
      for (int r = 0; r < 8; ++r)
        lds_p[m][ss * 16 + r + 8 * h] = (_Float16)pv[ss][r];
    __syncthreads();
    // B fragment of P^T: lane column t = m, K = s: lanes 0-15 s{0..7,16..23}, 16-31 s{8..15,24..31}
    v16h pb = cat16(*(const v8h*)(&lds_p[m][h * 8]),
                    *(const v8h*)(&lds_p[m][16 + h * 8]));
    __syncthreads();

    // ---- O^T += V^T_tile(16d x 32s) * P^T(32s x 16t) per d-tile ----
    v16h va[4];
#pragma unroll
    for (int nt = 0; nt < 4; ++nt) {
      const _Float16* vr = vt + ((size_t)b * D_ + nt * 16 + m) * T_ + s0;
      va[nt] = cat16(*(const v8h*)(vr + h * 8), *(const v8h*)(vr + 16 + h * 8));
    }
#pragma unroll
    for (int nt = 0; nt < 4; ++nt)
      o[nt] = __builtin_amdgcn_wmma_f32_16x16x32_f16(
          false, va[nt], false, pb, (short)0, o[nt], false, false);
  }

  // ---- epilogue: O^T C-layout -> out[t][d]; consecutive r = consecutive d ----
  const float inv = 1.0f / lrun;
  float* orow = out + ((size_t)(b * T_ + trow)) * D_;
#pragma unroll
  for (int nt = 0; nt < 4; ++nt) {
    v4f lo, hi;
#pragma unroll
    for (int i = 0; i < 4; ++i) { lo[i] = o[nt][i] * inv; hi[i] = o[nt][4 + i] * inv; }
    // d base = nt*16 + 8h : this lane's 8 consecutive d values
    *(v4f*)(orow + nt * 16 + 8 * h)     = lo;
    *(v4f*)(orow + nt * 16 + 8 * h + 4) = hi;
  }
}

extern "C" void kernel_launch(void* const* d_in, const int* in_sizes, int n_in,
                              void* d_out, int out_size, void* d_ws, size_t ws_size,
                              hipStream_t stream) {
  (void)in_sizes; (void)n_in; (void)out_size; (void)ws_size;
  const float* x  = (const float*)d_in[0];
  const float* Wq = (const float*)d_in[1];
  const float* Wk = (const float*)d_in[2];
  const float* Wv = (const float*)d_in[3];
  float* out = (float*)d_out;

  const size_t WT_BYTES  = (size_t)3 * D_ * C_ * sizeof(_Float16);      // 147456
  const size_t QKV_BYTES = (size_t)B_ * T_ * D_ * sizeof(_Float16);     // 4 MiB each
  _Float16* wt = (_Float16*)d_ws;
  _Float16* qh = (_Float16*)((char*)d_ws + WT_BYTES);
  _Float16* kh = (_Float16*)((char*)d_ws + WT_BYTES + QKV_BYTES);
  _Float16* vt = (_Float16*)((char*)d_ws + WT_BYTES + 2 * QKV_BYTES);

  prep_w<<<(3 * C_ * D_ + 255) / 256, 256, 0, stream>>>(Wq, Wk, Wv, wt);
  proj_qkv<<<B_ * (T_ / 16), 32, 0, stream>>>(x, wt, qh, kh, vt);
  attn<<<B_ * (T_ / 16), 32, 0, stream>>>(qh, kh, vt, out);
}
